// Transformer_60266981097602
// MI455X (gfx1250) — matrix operand order repaired and checked by reading, not compiled, not run on hardware
//
#include <hip/hip_runtime.h>
#include <hip/hip_bf16.h>

// ---------------------------------------------------------------------------
// MI455X (gfx1250) transformer forward. wave32; all matmuls via
// v_wmma_f32_16x16x32_f16 (fp32 operands converted to f16 in LDS staging,
// fp32 accumulate). Flash-style attention (no S^2 scores buffer).
// Round 3: BK=64 (16 WMMAs per staging round, half the barrier overhead),
// register-level pipelining (next tile's float4 loads in flight under the
// WMMA stream), templated bT/ReLU to remove hot-loop branches.
// ---------------------------------------------------------------------------

typedef __attribute__((ext_vector_type(16))) _Float16 v16h;
typedef __attribute__((ext_vector_type(8)))  _Float16 v8h;
typedef __attribute__((ext_vector_type(4)))  _Float16 v4h;
typedef __attribute__((ext_vector_type(8)))  float    v8f;

#define WMMA_F32_F16(a, b, c) \
    __builtin_amdgcn_wmma_f32_16x16x32_f16(false, (a), false, (b), (short)0, (c), false, false)

#define SHUF16(lo, hi) __builtin_shufflevector((lo), (hi), 0, 1, 2, 3, 4, 5, 6, 7, \
                                               8, 9, 10, 11, 12, 13, 14, 15)

__device__ __forceinline__ v4h cvt4(const float4& t) {
    v4h c;
    c[0] = (_Float16)t.x; c[1] = (_Float16)t.y;
    c[2] = (_Float16)t.z; c[3] = (_Float16)t.w;
    return c;
}

// ---------------------------------------------------------------------------
// Embedding: x[t, :] = emb[tokens[t], :] + pos_emb[t % S, :]
// ---------------------------------------------------------------------------
__global__ __launch_bounds__(256) void embed_kernel(
    const int* __restrict__ tokens, const float* __restrict__ emb,
    const float* __restrict__ pos, float* __restrict__ x)
{
    constexpr int D = 1024, S = 1024;
    const int t    = blockIdx.x;      // flat token index in [0, B*S)
    const int sidx = t & (S - 1);
    const int tok  = tokens[t];
    const int d    = threadIdx.x * 4;
    const float4 e = *(const float4*)&emb[(size_t)tok * D + d];
    const float4 p = *(const float4*)&pos[(size_t)sidx * D + d];
    float4 r; r.x = e.x + p.x; r.y = e.y + p.y; r.z = e.z + p.z; r.w = e.w + p.w;
    *(float4*)&x[(size_t)t * D + d] = r;
}

// ---------------------------------------------------------------------------
// GEMM: C[M,N] = A[M,K] * B + bias (optional), optional ReLU.
//   BT == 0 : B is [K,N] row-major (x @ W)
//   BT == 1 : B is [N,K] row-major (x @ emb^T, tied head)
// Block tile 128x128x64, 256 threads = 8 waves (2x4); each wave owns a
// 64x32 sub-tile = 4x2 WMMA accumulators, 2 K-substeps -> 16 WMMAs per
// staging round.  Single LDS buffer; pipelining is register-level: the
// next tile's global float4 loads are issued before computing the current
// tile from LDS, hiding HBM latency under the WMMA stream.
// ---------------------------------------------------------------------------
template<int BT, int RELU>
__global__ __launch_bounds__(256) void gemm_wmma_f16(
    const float* __restrict__ A, const float* __restrict__ Bm,
    const float* __restrict__ bias, float* __restrict__ C,
    int M, int N, int K)
{
    constexpr int BM = 128, BN = 128, BK = 64, BKP = 72; // 144B LDS row stride
    __shared__ __attribute__((aligned(16))) _Float16 As[BM * BKP]; // 18.4 KB
    __shared__ __attribute__((aligned(16))) _Float16 Bs[BN * BKP]; // 18.4 KB

    const int m0  = blockIdx.y * BM;
    const int n0  = blockIdx.x * BN;
    const int tid = threadIdx.x;
    const int lane = tid & 31;
    const int w    = tid >> 5;
    const int wr   = w >> 2;          // 0..1 -> 64 rows each
    const int wc   = w & 3;           // 0..3 -> 32 cols each
    const int mloc = lane & 15;
    const bool hiHalf = lane >= 16;
    const int kbA = hiHalf ? 8 : 0;   // A frag K-base (ISA 16-bit A layout)
    const int kbB = hiHalf ? 8 : 0;   // B frag K-base (same K order as A)

    float4 aReg[8], bReg[8];

    // gather next K-tile into registers: all global loads independent/in-flight
    auto loadRegs = [&](int k0) {
#pragma unroll
        for (int it = 0; it < 8; ++it) {
            const int f  = tid + it * 256;       // float4 index, 0..2047
            const int mm = f >> 4, kk = (f & 15) * 4;
            aReg[it] = *(const float4*)&A[(size_t)(m0 + mm) * K + k0 + kk];
        }
#pragma unroll
        for (int it = 0; it < 8; ++it) {
            const int f = tid + it * 256;
            if (BT) {
                const int nn = f >> 4, kk = (f & 15) * 4;
                bReg[it] = *(const float4*)&Bm[(size_t)(n0 + nn) * K + k0 + kk];
            } else {
                const int kk = f >> 5, nn = (f & 31) * 4;
                bReg[it] = *(const float4*)&Bm[(size_t)(k0 + kk) * N + n0 + nn];
            }
        }
    };
    // convert & commit registers -> LDS
    auto storeRegs = [&]() {
#pragma unroll
        for (int it = 0; it < 8; ++it) {
            const int f  = tid + it * 256;
            const int mm = f >> 4, kk = (f & 15) * 4;
            *(v4h*)&As[mm * BKP + kk] = cvt4(aReg[it]);   // ds_store_b64
        }
#pragma unroll
        for (int it = 0; it < 8; ++it) {
            const int f = tid + it * 256;
            if (BT) {
                const int nn = f >> 4, kk = (f & 15) * 4;
                *(v4h*)&Bs[nn * BKP + kk] = cvt4(bReg[it]);
            } else {  // transpose through registers: 4 scattered b16 stores
                const int kk = f >> 5, nn = (f & 31) * 4;
                Bs[(nn + 0) * BKP + kk] = (_Float16)bReg[it].x;
                Bs[(nn + 1) * BKP + kk] = (_Float16)bReg[it].y;
                Bs[(nn + 2) * BKP + kk] = (_Float16)bReg[it].z;
                Bs[(nn + 3) * BKP + kk] = (_Float16)bReg[it].w;
            }
        }
    };

    v8f acc[4][2] = {};
    const int nk = K / BK;

    loadRegs(0);
    storeRegs();
    __syncthreads();

    for (int kt = 0; kt < nk; ++kt) {
        if (kt + 1 < nk) {
            loadRegs((kt + 1) * BK);             // in flight across compute
            if (kt + 2 < nk) {                   // L2 prefetch 2 tiles ahead
                const int kp = (kt + 2) * BK;
                __builtin_prefetch(&A[(size_t)(m0 + (tid >> 1)) * K + kp + (tid & 1) * 32], 0, 3);
                if (BT)
                    __builtin_prefetch(&Bm[(size_t)(n0 + (tid >> 1)) * K + kp + (tid & 1) * 32], 0, 3);
                else
                    __builtin_prefetch(&Bm[(size_t)(kp + (tid >> 2)) * N + n0 + (tid & 3) * 32], 0, 3);
            }
        }
        // ---- 2 K-substeps: fragments from LDS + 16 WMMAs total ----
#pragma unroll
        for (int ks = 0; ks < 2; ++ks) {
            const int ko = ks * 32;
            v16h af[4];
#pragma unroll
            for (int tr = 0; tr < 4; ++tr) {
                const int mm = wr * 64 + tr * 16 + mloc;
                v8h lo = *(const v8h*)&As[mm * BKP + ko + kbA];
                v8h hi = *(const v8h*)&As[mm * BKP + ko + kbA + 16];
                af[tr] = SHUF16(lo, hi);
            }
            v16h bf[2];
#pragma unroll
            for (int tc = 0; tc < 2; ++tc) {
                const int nn = wc * 32 + tc * 16 + mloc;
                v8h lo = *(const v8h*)&Bs[nn * BKP + ko + kbB];
                v8h hi = *(const v8h*)&Bs[nn * BKP + ko + kbB + 16];
                bf[tc] = SHUF16(lo, hi);
            }
#pragma unroll
            for (int tr = 0; tr < 4; ++tr)
#pragma unroll
                for (int tc = 0; tc < 2; ++tc)
                    acc[tr][tc] = WMMA_F32_F16(af[tr], bf[tc], acc[tr][tc]);
        }

        __syncthreads();                          // everyone done reading LDS
        if (kt + 1 < nk) {
            storeRegs();
            __syncthreads();                      // stores visible
        }
    }

    // ---- epilogue: bias (+ReLU), stores per WMMA C-tile layout ----
#pragma unroll
    for (int tr = 0; tr < 4; ++tr) {
#pragma unroll
        for (int tc = 0; tc < 2; ++tc) {
            const int gc = n0 + wc * 32 + tc * 16 + mloc;
            const float bb = bias ? bias[gc] : 0.0f;
#pragma unroll
            for (int r = 0; r < 8; ++r) {
                const int gr = m0 + wr * 64 + tr * 16 + r + (hiHalf ? 8 : 0);
                float v = acc[tr][tc][r] + bb;
                if (RELU) v = fmaxf(v, 0.0f);
                C[(size_t)gr * N + gc] = v;
            }
        }
    }
}

// ---------------------------------------------------------------------------
// Flash attention, one wave (32 threads) per 16-row Q tile per (b, h).
// hd = 64.  Scores via 2 WMMAs (K-dim 64 = 2x32); keys in steps of 32 (one
// WMMA-K for P@V); online softmax with shfl_xor reductions across the
// 16-lane halves; P staged through LDS as f16.
// ---------------------------------------------------------------------------
__global__ __launch_bounds__(32) void attn_flash(
    const float* __restrict__ Q, const float* __restrict__ Kb,
    const float* __restrict__ Vb, float* __restrict__ O)
{
    constexpr int D = 1024, S = 1024, H = 16;
    __shared__ __attribute__((aligned(16))) _Float16 P[16 * 32];

    const int qt   = blockIdx.x;          // q tile (16 rows)
    const int b    = blockIdx.y >> 4;     // / H
    const int h    = blockIdx.y & (H - 1);
    const int lane = threadIdx.x;
    const int mloc = lane & 15;
    const bool hiHalf = lane >= 16;
    const int kbA = hiHalf ? 8 : 0;
    const int kb2 = hiHalf ? 16 : 0;
    const float scale = 0.125f;           // 1/sqrt(64)

    const float* qp = Q  + (size_t)b * S * D + h * 64;
    const float* kp = Kb + (size_t)b * S * D + h * 64;
    const float* vp = Vb + (size_t)b * S * D + h * 64;

    // Q A-fragments (16x32 each), scale folded in before f16 conversion
    const int qrow = qt * 16 + mloc;
    v16h qa[2];
#pragma unroll
    for (int f = 0; f < 2; ++f) {
        const float* r = qp + (size_t)qrow * D + f * 32 + kbA;
        const float4 t0 = *(const float4*)(r);
        const float4 t1 = *(const float4*)(r + 4);
        const float4 t2 = *(const float4*)(r + 16);
        const float4 t3 = *(const float4*)(r + 20);
        qa[f][0]  = (_Float16)(t0.x * scale); qa[f][1]  = (_Float16)(t0.y * scale);
        qa[f][2]  = (_Float16)(t0.z * scale); qa[f][3]  = (_Float16)(t0.w * scale);
        qa[f][4]  = (_Float16)(t1.x * scale); qa[f][5]  = (_Float16)(t1.y * scale);
        qa[f][6]  = (_Float16)(t1.z * scale); qa[f][7]  = (_Float16)(t1.w * scale);
        qa[f][8]  = (_Float16)(t2.x * scale); qa[f][9]  = (_Float16)(t2.y * scale);
        qa[f][10] = (_Float16)(t2.z * scale); qa[f][11] = (_Float16)(t2.w * scale);
        qa[f][12] = (_Float16)(t3.x * scale); qa[f][13] = (_Float16)(t3.y * scale);
        qa[f][14] = (_Float16)(t3.z * scale); qa[f][15] = (_Float16)(t3.w * scale);
    }

    float mrun[8], lrun[8];
#pragma unroll
    for (int r = 0; r < 8; ++r) { mrun[r] = -3.0e38f; lrun[r] = 0.0f; }
    v8f o[4] = {};

    const int myrow = qt * 16 + (hiHalf ? 8 : 0); // this lane-half's row base
    const int npair = (qt * 16 + 16 + 31) >> 5;   // #32-key steps (causal)

    for (int p = 0; p < npair; ++p) {
        // ---- scores for 32 keys: two 16x16 C tiles ----
        v8f sc[2] = {};
#pragma unroll
        for (int sub = 0; sub < 2; ++sub) {
            const int key = p * 32 + sub * 16 + mloc;
#pragma unroll
            for (int f = 0; f < 2; ++f) {
                const float* kr = kp + (size_t)key * D + f * 32 + kbA;
                const float4 t0 = *(const float4*)(kr);
                const float4 t1 = *(const float4*)(kr + 4);
                const float4 t2 = *(const float4*)(kr + 16);
                const float4 t3 = *(const float4*)(kr + 20);
                v16h kf;
                kf[0]  = (_Float16)t0.x; kf[1]  = (_Float16)t0.y;
                kf[2]  = (_Float16)t0.z; kf[3]  = (_Float16)t0.w;
                kf[4]  = (_Float16)t1.x; kf[5]  = (_Float16)t1.y;
                kf[6]  = (_Float16)t1.z; kf[7]  = (_Float16)t1.w;
                kf[8]  = (_Float16)t2.x; kf[9]  = (_Float16)t2.y;
                kf[10] = (_Float16)t2.z; kf[11] = (_Float16)t2.w;
                kf[12] = (_Float16)t3.x; kf[13] = (_Float16)t3.y;
                kf[14] = (_Float16)t3.z; kf[15] = (_Float16)t3.w;
                sc[sub] = WMMA_F32_F16(qa[f], kf, sc[sub]);
            }
        }
        // ---- causal mask ----
#pragma unroll
        for (int sub = 0; sub < 2; ++sub) {
            const int keyg = p * 32 + sub * 16 + mloc;
#pragma unroll
            for (int r = 0; r < 8; ++r)
                if (keyg > myrow + r) sc[sub][r] = -3.0e38f;
        }
        // ---- online softmax (row stats across 16-lane halves) ----
        float ps0[8], ps1[8];
#pragma unroll
        for (int r = 0; r < 8; ++r) {
            float rv = fmaxf(sc[0][r], sc[1][r]);
#pragma unroll
            for (int off = 8; off > 0; off >>= 1)
                rv = fmaxf(rv, __shfl_xor(rv, off, 16));
            const float nm = fmaxf(mrun[r], rv);
            const float a  = __expf(mrun[r] - nm);
            const float p0 = __expf(sc[0][r] - nm);
            const float p1 = __expf(sc[1][r] - nm);
            float rs = p0 + p1;
#pragma unroll
            for (int off = 8; off > 0; off >>= 1)
                rs += __shfl_xor(rs, off, 16);
            lrun[r] = lrun[r] * a + rs;
            mrun[r] = nm;
#pragma unroll
            for (int d = 0; d < 4; ++d) o[d][r] *= a;
            ps0[r] = p0; ps1[r] = p1;
        }
        // ---- stage P (16x32 f16) in LDS, reload as A fragment ----
        __syncthreads();
#pragma unroll
        for (int r = 0; r < 8; ++r) {
            const int rl = r + (hiHalf ? 8 : 0);
            P[rl * 32 + mloc]      = (_Float16)ps0[r];
            P[rl * 32 + 16 + mloc] = (_Float16)ps1[r];
        }
        __syncthreads();
        v8h plo = *(const v8h*)&P[mloc * 32 + kbA];
        v8h phi = *(const v8h*)&P[mloc * 32 + kbA + 16];
        v16h pa = SHUF16(plo, phi);
        // ---- O += P @ V (4 output d-blocks of 16) ----
#pragma unroll
        for (int d = 0; d < 4; ++d) {
            float tv[16];
#pragma unroll
            for (int i = 0; i < 16; ++i)   // independent, stay in flight
                tv[i] = vp[(size_t)(p * 32 + kbA + (i < 8 ? i : i + 8)) * D + d * 16 + mloc];
            v16h vf;
#pragma unroll
            for (int i = 0; i < 16; ++i)
                vf[i] = (_Float16)tv[i];
            o[d] = WMMA_F32_F16(pa, vf, o[d]);
        }
    }

    // ---- normalize and write ctx ----
    float* op = O + (size_t)b * S * D + h * 64;
#pragma unroll
    for (int r = 0; r < 8; ++r) {
        const float inv = 1.0f / lrun[r];
        const int grow = myrow + r;
#pragma unroll
        for (int d = 0; d < 4; ++d)
            op[(size_t)grow * D + d * 16 + mloc] = o[d][r] * inv;
    }
}

// ---------------------------------------------------------------------------
// out[row,:] = LayerNorm(x[row,:] + y[row,:]) * g + b   (one block per row)
// ---------------------------------------------------------------------------
__global__ __launch_bounds__(256) void add_layernorm(
    const float* __restrict__ x, const float* __restrict__ y,
    const float* __restrict__ g, const float* __restrict__ bb,
    float* __restrict__ out)
{
    constexpr int D = 1024;
    __shared__ float red[256];
    const int row = blockIdx.x, tid = threadIdx.x;
    const int d = tid * 4;
    const float4 xv = *(const float4*)&x[(size_t)row * D + d];
    const float4 yv = *(const float4*)&y[(size_t)row * D + d];
    float v[4] = { xv.x + yv.x, xv.y + yv.y, xv.z + yv.z, xv.w + yv.w };
    float s = v[0] + v[1] + v[2] + v[3];
    red[tid] = s; __syncthreads();
    for (int st = 128; st > 0; st >>= 1) {
        if (tid < st) red[tid] += red[tid + st];
        __syncthreads();
    }
    const float mu = red[0] * (1.0f / D);
    __syncthreads();
    float sq = 0.0f;
#pragma unroll
    for (int i = 0; i < 4; ++i) { const float dv = v[i] - mu; sq += dv * dv; }
    red[tid] = sq; __syncthreads();
    for (int st = 128; st > 0; st >>= 1) {
        if (tid < st) red[tid] += red[tid + st];
        __syncthreads();
    }
    const float rstd = rsqrtf(red[0] * (1.0f / D) + 1e-5f);
    const float4 gv = *(const float4*)&g[d];
    const float4 bv = *(const float4*)&bb[d];
    float4 ov;
    ov.x = gv.x * (v[0] - mu) * rstd + bv.x;
    ov.y = gv.y * (v[1] - mu) * rstd + bv.y;
    ov.z = gv.z * (v[2] - mu) * rstd + bv.z;
    ov.w = gv.w * (v[3] - mu) * rstd + bv.w;
    *(float4*)&out[(size_t)row * D + d] = ov;
}

// ---------------------------------------------------------------------------
// Host launcher
// ---------------------------------------------------------------------------
extern "C" void kernel_launch(void* const* d_in, const int* in_sizes, int n_in,
                              void* d_out, int out_size, void* d_ws, size_t ws_size,
                              hipStream_t stream) {
    (void)in_sizes; (void)n_in; (void)out_size; (void)ws_size;
    constexpr int L = 8, V = 32000, D = 1024, B = 4, S = 1024, FF = 4096;
    constexpr int M = B * S;   // 4096 tokens
    constexpr int H = 16;

    const int*   tokens = (const int*)  d_in[0];
    // d_in[1] = num_heads (compile-time constant H here)
    const float* emb    = (const float*)d_in[2];
    const float* pos    = (const float*)d_in[3];
    const float* Wq = (const float*)d_in[4];  const float* bq = (const float*)d_in[5];
    const float* Wk = (const float*)d_in[6];  const float* bk = (const float*)d_in[7];
    const float* Wv = (const float*)d_in[8];  const float* bv = (const float*)d_in[9];
    const float* Wo = (const float*)d_in[10]; const float* bo = (const float*)d_in[11];
    const float* n1s = (const float*)d_in[12]; const float* n1b = (const float*)d_in[13];
    const float* W1 = (const float*)d_in[14]; const float* b1 = (const float*)d_in[15];
    const float* W2 = (const float*)d_in[16]; const float* b2 = (const float*)d_in[17];
    const float* n2s = (const float*)d_in[18]; const float* n2b = (const float*)d_in[19];

    // workspace layout (bytes): 6 * M*D*4 + M*FF*4  ~= 168 MB
    char* ws = (char*)d_ws;
    const size_t szMD = (size_t)M * D * sizeof(float);
    float* x  = (float*)(ws + 0 * szMD);
    float* qb = (float*)(ws + 1 * szMD);
    float* kb = (float*)(ws + 2 * szMD);
    float* vb = (float*)(ws + 3 * szMD);
    float* cb = (float*)(ws + 4 * szMD);   // attention context
    float* yb = (float*)(ws + 5 * szMD);   // generic GEMM output
    float* hb = (float*)(ws + 6 * szMD);   // FFN hidden, M*FF

    embed_kernel<<<M, 256, 0, stream>>>(tokens, emb, pos, x);

    const dim3 blk(256);
    const dim3 gD(D / 128, M / 128);
    const dim3 gF(FF / 128, M / 128);
    const dim3 gV(V / 128, M / 128);
    const dim3 gA(S / 16, B * H);

    for (int l = 0; l < L; ++l) {
        const size_t wofs = (size_t)l * D * D;
        const size_t fofs = (size_t)l * D * FF;
        gemm_wmma_f16<0, 0><<<gD, blk, 0, stream>>>(x, Wq + wofs, bq + l * D, qb, M, D, D);
        gemm_wmma_f16<0, 0><<<gD, blk, 0, stream>>>(x, Wk + wofs, bk + l * D, kb, M, D, D);
        gemm_wmma_f16<0, 0><<<gD, blk, 0, stream>>>(x, Wv + wofs, bv + l * D, vb, M, D, D);
        attn_flash<<<gA, dim3(32), 0, stream>>>(qb, kb, vb, cb);
        gemm_wmma_f16<0, 0><<<gD, blk, 0, stream>>>(cb, Wo + wofs, bo + l * D, yb, M, D, D);
        add_layernorm<<<M, 256, 0, stream>>>(x, yb, n1s + l * D, n1b + l * D, x);
        gemm_wmma_f16<0, 1><<<gF, blk, 0, stream>>>(x, W1 + fofs, b1 + l * FF, hb, M, FF, D);
        gemm_wmma_f16<0, 0><<<gD, blk, 0, stream>>>(hb, W2 + fofs, b2 + l * D, yb, M, D, FF);
        add_layernorm<<<M, 256, 0, stream>>>(x, yb, n2s + l * D, n2b + l * D, x);
    }
    // tied output head: logits = x @ emb^T   (B is [V, D] row-major -> bT=1)
    gemm_wmma_f16<1, 0><<<gV, blk, 0, stream>>>(x, emb, nullptr, (float*)d_out, M, V, D);
}
